// S2Attention_83700322665218
// MI455X (gfx1250) — compile-verified, matrix operand order checked by reading, not run on hardware
//
#include <hip/hip_runtime.h>
#include <hip/hip_bf16.h>

// ---------- CDNA5 feature detection ----------------------------------------
#if defined(__has_builtin)
#if __has_builtin(__builtin_amdgcn_global_load_async_to_lds_b128) && \
    __has_builtin(__builtin_amdgcn_s_wait_asynccnt)
#define HAVE_ASYNC_LDS 1
#endif
#endif

// ---------- types / helpers -------------------------------------------------
typedef __attribute__((ext_vector_type(16))) __bf16 v16bf;
typedef __attribute__((ext_vector_type(8)))  float  v8f;
typedef int v4i __attribute__((vector_size(16)));   // async builtin's pointee type

union Frag { v16bf v; uint4 q[2]; };

__device__ inline unsigned short f2bf(float f) {   // native v_cvt / v_fma_mix path
  union { __bf16 h; unsigned short s; } u;
  u.h = (__bf16)f;
  return u.s;
}
__device__ inline float bf2f(unsigned short s) {
  union { __bf16 h; unsigned short s; } u;
  u.s = s;
  return (float)u.h;
}
__device__ inline v8f zero8() {
  v8f z = {0.f,0.f,0.f,0.f,0.f,0.f,0.f,0.f};
  return z;
}
__device__ inline v8f wmma_bf16(v16bf a, v16bf b, v8f c) {
  return __builtin_amdgcn_wmma_f32_16x16x32_bf16(
      false, a, false, b, (short)0, c, false, false);
}

// problem constants
#define BB 8
#define CC 512
#define NQ 4096     // 64*64
#define NK 289      // 17*17
#define NKP 304     // 19*16 padded K rows
#define NKP2 320    // 10*32 padded for P@V K loop
#define DD 64
#define HH 8

// ---------- K0: weights f32 -> bf16 (same [co][cin] layout = WMMA-B-ready) --
__global__ void convert_weights_kernel(const float* __restrict__ wq,
                                       const float* __restrict__ wpw,
                                       const float* __restrict__ wout,
                                       unsigned short* __restrict__ wb) {
  int idx = blockIdx.x * blockDim.x + threadIdx.x;
  if (idx >= 3 * CC * CC) return;
  int m = idx >> 18;                 // which matrix
  int r = idx & (CC * CC - 1);
  const float* src = (m == 0) ? wq : (m == 1) ? wpw : wout;
  wb[idx] = f2bf(src[r]);
}

// ---------- K1: sem [B][C][4096] f32 -> semT [B][4096][C] bf16 --------------
__global__ void __launch_bounds__(256)
transpose_bf16_kernel(const float* __restrict__ in, unsigned short* __restrict__ out) {
  __shared__ float tile[32][33];
  const int b = blockIdx.z;
  const int p0 = blockIdx.x * 32, c0 = blockIdx.y * 32;
  const int tx = threadIdx.x, ty = threadIdx.y;     // (32, 8)
#pragma unroll
  for (int i = 0; i < 4; ++i) {
    const int c = c0 + ty + i * 8;
    tile[ty + i * 8][tx] = in[((size_t)(b * CC + c)) * NQ + p0 + tx];
  }
  __syncthreads();
#pragma unroll
  for (int i = 0; i < 4; ++i) {
    const int p = p0 + ty + i * 8;
    out[((size_t)(b * NQ + p)) * CC + c0 + tx] = f2bf(tile[tx][ty + i * 8]);
  }
}

// ---------- K2: depthwise 4x4 stride-4 pad-2 conv + channel LayerNorm -------
__global__ void __launch_bounds__(256)
dwconv_ln_kernel(const float* __restrict__ in,   // [B][512][64][64]
                 const float* __restrict__ wdw,  // [512][4][4]
                 const float* __restrict__ g,
                 const float* __restrict__ beta,
                 unsigned short* __restrict__ out) {  // bf16 [B][289][512]
  __shared__ float ssum[256], ssq[256];
  const int p = blockIdx.x, b = blockIdx.y;
  const int oy = p / 17, ox = p % 17;
  const int t = threadIdx.x;
  float v[2];
#pragma unroll
  for (int i = 0; i < 2; ++i) {
    const int c = t + i * 256;
    const float* ip = in + ((size_t)(b * CC + c)) * NQ;
    const float* wp = wdw + c * 16;
    float acc = 0.f;
#pragma unroll
    for (int ky = 0; ky < 4; ++ky) {
      const int y = oy * 4 - 2 + ky;
      if ((unsigned)y < 64u) {
#pragma unroll
        for (int kx = 0; kx < 4; ++kx) {
          const int x = ox * 4 - 2 + kx;
          if ((unsigned)x < 64u) acc += ip[y * 64 + x] * wp[ky * 4 + kx];
        }
      }
    }
    v[i] = acc;
  }
  ssum[t] = v[0] + v[1];
  ssq[t]  = v[0] * v[0] + v[1] * v[1];
  __syncthreads();
  for (int off = 128; off > 0; off >>= 1) {
    if (t < off) { ssum[t] += ssum[t + off]; ssq[t] += ssq[t + off]; }
    __syncthreads();
  }
  const float mean = ssum[0] * (1.f / 512.f);
  const float var  = ssq[0] * (1.f / 512.f) - mean * mean;
  const float rstd = rsqrtf(var + 1e-5f);
#pragma unroll
  for (int i = 0; i < 2; ++i) {
    const int c = t + i * 256;
    out[((size_t)(b * NK + p)) * CC + c] = f2bf((v[i] - mean) * rstd * g[c] + beta[c]);
  }
}

// ---------- K3: tiled bf16 WMMA GEMM:  out[b] = A[b](Mx512) * W^T ----------
// W given bf16 [co][cin] (row-major) == column-major B operand, read direct.
// A tile double-buffered in LDS via async global->LDS when tile is full.
// F32T==0: bf16 out [b][m][512];  F32T==1: f32 out [b][n][4096] (transposed)
template <int F32T>
__global__ void __launch_bounds__(128)
gemm512_kernel(const unsigned short* __restrict__ A, size_t aBatch,
               int Mvalid, int Mstore,
               const unsigned short* __restrict__ Wb,
               unsigned short* __restrict__ obf, float* __restrict__ of32,
               size_t oBatch) {
  __shared__ __align__(16) unsigned short As[2][64][32];
  const int b = blockIdx.z;
  const int mbase = blockIdx.y * 64, nbase = blockIdx.x * 64;
  const int t = threadIdx.x, wave = t >> 5, lane = t & 31;
  const int lo = lane & 15, hi = lane >> 4;
  const unsigned short* Ab = A + (size_t)b * aBatch;
  const bool fullTile = (mbase + 64) <= Mvalid;
  (void)fullTile;

  auto stageA = [&](int buf, int kk) {
#pragma unroll
    for (int i = 0; i < 2; ++i) {
      const int cid = t + i * 128, row = cid >> 2, ch = cid & 3;
      const int m = mbase + row;
#ifdef HAVE_ASYNC_LDS
      if (fullTile) {
        __builtin_amdgcn_global_load_async_to_lds_b128(
            (v4i*)(Ab + (size_t)m * CC + kk + ch * 8),
            (v4i*)&As[buf][row][ch * 8], 0, 0);
      } else
#endif
      {
        uint4 val = {0u, 0u, 0u, 0u};
        if (m < Mvalid) {
          const unsigned short* src = Ab + (size_t)m * CC + kk + ch * 8;
          val = *(const uint4*)src;
          if (kk + 32 < 512) __builtin_prefetch(src + 32, 0, 0);  // global_prefetch_b8
        }
        *(uint4*)&As[buf][row][ch * 8] = val;
      }
    }
  };

  v8f acc[4];
#pragma unroll
  for (int i = 0; i < 4; ++i) acc[i] = zero8();

  stageA(0, 0);
#ifdef HAVE_ASYNC_LDS
  if (fullTile) __builtin_amdgcn_s_wait_asynccnt(0);
#endif
  __syncthreads();

  int buf = 0;
  for (int kk = 0; kk < 512; kk += 32) {
    if (kk + 32 < 512) stageA(buf ^ 1, kk + 32);   // prefetch next tile (async)

    Frag af;
    af.q[0] = *(const uint4*)&As[buf][wave * 16 + lo][hi * 8];
    af.q[1] = *(const uint4*)&As[buf][wave * 16 + lo][hi * 8 + 16];
#pragma unroll
    for (int nt = 0; nt < 4; ++nt) {
      const unsigned short* wrow =
          Wb + (size_t)(nbase + nt * 16 + lo) * CC + kk + hi * 8;
      Frag bfr;
      bfr.q[0] = *(const uint4*)wrow;
      bfr.q[1] = *(const uint4*)(wrow + 16);
      acc[nt] = wmma_bf16(af.v, bfr.v, acc[nt]);
    }

#ifdef HAVE_ASYNC_LDS
    if (fullTile) __builtin_amdgcn_s_wait_asynccnt(0);  // own ASYNCcnt, then barrier
#endif
    __syncthreads();
    buf ^= 1;
  }

  // epilogue: C/D layout M = r + 8*hi, N = lo
#pragma unroll
  for (int nt = 0; nt < 4; ++nt) {
#pragma unroll
    for (int r = 0; r < 8; ++r) {
      const int m = mbase + wave * 16 + r + 8 * hi;
      const int n = nbase + nt * 16 + lo;
      if (m < Mstore) {
        if (F32T) of32[(size_t)b * oBatch + (size_t)n * NQ + m] = acc[nt][r];
        else      obf[(size_t)b * oBatch + (size_t)m * CC + n] = f2bf(acc[nt][r]);
      }
    }
  }
}

// ---------- K4: attention per (b,h): S = scale*Q*K^T, softmax, O = P*V ------
__global__ void __launch_bounds__(64)
attn_kernel(const unsigned short* __restrict__ qb,
            const unsigned short* __restrict__ kb,
            const unsigned short* __restrict__ vb,
            unsigned short* __restrict__ ao) {
  __shared__ __align__(16) unsigned short Vt[DD][NKP2];    // V^T [dc][j]
  __shared__ __align__(16) unsigned short Sb[2][16][NKP2]; // per-wave scores/probs
  __shared__ float rowinv[2][16];

  const int bh = blockIdx.x;
  const int b = bh >> 3, h = bh & 7;
  const int t = threadIdx.x;
  const int wave = t >> 5, lane = t & 31, lo = lane & 15, hi = lane >> 4;
  const int qrow0 = blockIdx.y * 32 + wave * 16;

  // stage V^T (coalesced reads: 64 lanes = 64 consecutive dc)
  {
    const unsigned short* vbase = vb + (size_t)(b * NKP) * CC + h * DD + t;
    for (int j = 0; j < NKP2; ++j)
      Vt[t][j] = (j < NKP) ? vbase[(size_t)j * CC] : (unsigned short)0;
  }
  __syncthreads();

  // Q fragments (K = 64 -> two 16x32 A-frags)
  Frag aq[2];
  {
    const size_t qoff = ((size_t)(b * NQ + qrow0 + lo)) * CC + h * DD + hi * 8;
#pragma unroll
    for (int s = 0; s < 2; ++s) {
      aq[s].q[0] = *(const uint4*)(qb + qoff + s * 32);
      aq[s].q[1] = *(const uint4*)(qb + qoff + s * 32 + 16);
    }
  }

  // S tiles: software-pipelined over jt; running pointer, stride 16*CC elems
  {
    const unsigned short* kp =
        kb + ((size_t)(b * NKP + lo)) * CC + h * DD + hi * 8;
    Frag c0, c1, n0, n1;
    c0.q[0] = *(const uint4*)(kp);
    c0.q[1] = *(const uint4*)(kp + 16);
    c1.q[0] = *(const uint4*)(kp + 32);
    c1.q[1] = *(const uint4*)(kp + 48);
    for (int jt = 0; jt < 19; ++jt) {
      const unsigned short* kn = kp + 16 * CC;   // next tile (16 rows ahead)
      if (jt < 18) {                             // prefetch next frags first
        n0.q[0] = *(const uint4*)(kn);
        n0.q[1] = *(const uint4*)(kn + 16);
        n1.q[0] = *(const uint4*)(kn + 32);
        n1.q[1] = *(const uint4*)(kn + 48);
      }
      v8f acc = zero8();
      acc = wmma_bf16(aq[0].v, c0.v, acc);
      acc = wmma_bf16(aq[1].v, c1.v, acc);
#pragma unroll
      for (int r = 0; r < 8; ++r)
        Sb[wave][r + 8 * hi][jt * 16 + lo] = f2bf(acc[r] * 0.125f); // scale d^-0.5
      c0 = n0; c1 = n1;
      kp = kn;
    }
  }
  __syncthreads();

  // masked softmax: 2 lanes per row
  {
    const int row = lo, half = hi;
    float mx = -3.0e38f;
    for (int j = half; j < NK; j += 2) mx = fmaxf(mx, bf2f(Sb[wave][row][j]));
    mx = fmaxf(mx, __shfl_xor(mx, 16, 32));
    float sum = 0.f;
    for (int j = half; j < NKP2; j += 2) {
      float p = 0.f;
      if (j < NK) { p = __expf(bf2f(Sb[wave][row][j]) - mx); sum += p; }
      Sb[wave][row][j] = f2bf(p);   // zeros padded columns too
    }
    sum += __shfl_xor(sum, 16, 32);
    if (half == 0) rowinv[wave][row] = 1.f / sum;
  }
  __syncthreads();

  // O = P * V  (K = 320 -> 10 steps, N = 64 -> 4 tiles)
  v8f o[4];
#pragma unroll
  for (int i = 0; i < 4; ++i) o[i] = zero8();
  for (int jt = 0; jt < 10; ++jt) {
    Frag pf;
    pf.q[0] = *(const uint4*)&Sb[wave][lo][jt * 32 + hi * 8];
    pf.q[1] = *(const uint4*)&Sb[wave][lo][jt * 32 + hi * 8 + 16];
#pragma unroll
    for (int nt = 0; nt < 4; ++nt) {
      Frag vf;
      vf.q[0] = *(const uint4*)&Vt[nt * 16 + lo][jt * 32 + hi * 8];
      vf.q[1] = *(const uint4*)&Vt[nt * 16 + lo][jt * 32 + hi * 8 + 16];
      o[nt] = wmma_bf16(pf.v, vf.v, o[nt]);
    }
  }

  // epilogue: fold 1/rowsum, write ao[b][p][h*64+dc]
#pragma unroll
  for (int nt = 0; nt < 4; ++nt) {
#pragma unroll
    for (int r = 0; r < 8; ++r) {
      const int m = r + 8 * hi;
      ao[((size_t)(b * NQ + qrow0 + m)) * CC + h * DD + nt * 16 + lo] =
          f2bf(o[nt][r] * rowinv[wave][m]);
    }
  }
}

// ---------- host-side orchestration ----------------------------------------
extern "C" void kernel_launch(void* const* d_in, const int* in_sizes, int n_in,
                              void* d_out, int out_size, void* d_ws, size_t ws_size,
                              hipStream_t stream) {
  const float* sem  = (const float*)d_in[0];
  const float* spa  = (const float*)d_in[1];
  const float* x    = (const float*)d_in[2];
  const float* wq   = (const float*)d_in[3];
  const float* wdw  = (const float*)d_in[4];
  const float* lng  = (const float*)d_in[5];
  const float* lnb  = (const float*)d_in[6];
  const float* wpw  = (const float*)d_in[7];
  const float* wout = (const float*)d_in[8];
  float* out = (float*)d_out;

  // workspace carve-out (bf16 everywhere), ~76 MB total
  char* ws = (char*)d_ws;
  unsigned short* wb   = (unsigned short*)ws;                        // 3*512*512
  unsigned short* semT = (unsigned short*)(ws + (size_t)3 * CC * CC * 2);
  unsigned short* qb   = semT + (size_t)BB * NQ * CC;
  unsigned short* tb   = qb   + (size_t)BB * NQ * CC;
  unsigned short* kb   = tb   + (size_t)BB * NK * CC;
  unsigned short* vb   = kb   + (size_t)BB * NKP * CC;
  unsigned short* ao   = semT;   // semT dead after q-GEMM; alias for attention out

  const unsigned short* wqB   = wb;
  const unsigned short* wpwB  = wb + CC * CC;
  const unsigned short* woutB = wb + 2 * CC * CC;

  convert_weights_kernel<<<3 * CC * CC / 256, 256, 0, stream>>>(wq, wpw, wout, wb);

  // q = wq * sem
  transpose_bf16_kernel<<<dim3(NQ / 32, CC / 32, BB), dim3(32, 8), 0, stream>>>(sem, semT);
  gemm512_kernel<0><<<dim3(8, NQ / 64, BB), 128, 0, stream>>>(
      semT, (size_t)NQ * CC, NQ, NQ, wqB, qb, nullptr, (size_t)NQ * CC);

  // k = pw * LN(dw(spa));  v = pw * LN(dw(x));  rows 289..303 zero-padded
  dwconv_ln_kernel<<<dim3(NK, BB), 256, 0, stream>>>(spa, wdw, lng, lnb, tb);
  gemm512_kernel<0><<<dim3(8, 5, BB), 128, 0, stream>>>(
      tb, (size_t)NK * CC, NK, NKP, wpwB, kb, nullptr, (size_t)NKP * CC);
  dwconv_ln_kernel<<<dim3(NK, BB), 256, 0, stream>>>(x, wdw, lng, lnb, tb);
  gemm512_kernel<0><<<dim3(8, 5, BB), 128, 0, stream>>>(
      tb, (size_t)NK * CC, NK, NKP, wpwB, vb, nullptr, (size_t)NKP * CC);

  // attention
  attn_kernel<<<dim3(BB * HH, NQ / 32), 64, 0, stream>>>(qb, kb, vb, ao);

  // out = wout * attn_out  (f32, NCHW layout via transposed store)
  gemm512_kernel<1><<<dim3(8, NQ / 64, BB), 128, 0, stream>>>(
      ao, (size_t)NQ * CC, NQ, NQ, woutB, nullptr, out, (size_t)CC * NQ);
}
